// GATGCNLSTM_75118978007586
// MI455X (gfx1250) — compile-verified, matrix operand order checked by reading, not run on hardware
//
#include <hip/hip_runtime.h>
#include <hip/hip_bf16.h>

#define N_NODES 20000
#define T_STEPS 8
#define E_EDGES 640000
#define F_IN    64
#define H_DIM   128
#define EN_EDGES (E_EDGES + N_NODES)
#define MT (N_NODES / 16)          // 1250 M-tiles of 16
#define KT_TOT 12                  // 384 K cols / 32

typedef __attribute__((ext_vector_type(16))) __bf16 v16bf;
typedef __attribute__((ext_vector_type(8)))  float  v8f;

// ---------- helpers ----------------------------------------------------------

// K position inside a 16x32 bf16 A-tile (and symmetric B layout) for value j
// of a lane, per CDNA5 ISA 7.12.2 (lanes 0-15: K=0..7,16..23; lanes 16-31: +8).
__device__ __forceinline__ int kpos(int j, int lane) {
  return (j < 8 ? j : j + 8) + ((lane & 16) ? 8 : 0);
}

__device__ __forceinline__ unsigned ordf(float f) {
  unsigned u = __float_as_uint(f);
  return (u & 0x80000000u) ? ~u : (u | 0x80000000u);
}
__device__ __forceinline__ float unordf(unsigned u) {
  u = (u & 0x80000000u) ? (u & 0x7FFFFFFFu) : ~u;
  return __uint_as_float(u);
}
__device__ __forceinline__ float sigmoidf(float x) {
  return 1.0f / (1.0f + __expf(-x));
}

// ---------- packing kernels --------------------------------------------------

// Pack f32 row-major [20000 x ld] columns [ktl*32 .. ) into bf16 A-operand
// tiles: Apk[(mt*ktTot + ktStart + ktl)*512 + lane*16 + j]
__global__ void pack_a_bf16(const float* __restrict__ src, int ld,
                            __bf16* __restrict__ Apk, int ktStart, int ktCount,
                            int ktTot) {
  int tid  = blockIdx.x * blockDim.x + threadIdx.x;
  int lane = tid & 31;
  int rest = tid >> 5;
  if (rest >= MT * ktCount) return;
  int ktl = rest % ktCount;
  int mt  = rest / ktCount;
  int m   = mt * 16 + (lane & 15);
  const float* srow = src + (size_t)m * ld + ktl * 32;
  __bf16* dstp = Apk + (((size_t)(mt * ktTot + ktStart + ktl)) << 9) + lane * 16;
#pragma unroll
  for (int j = 0; j < 16; ++j) dstp[j] = (__bf16)srow[kpos(j, lane)];
}

// Pack W_gat = [Wl | Wr] (64 x 128) into B tiles, layout [kt][nt] (Kt=2, Nt=8).
__global__ void pack_wgat(const float* __restrict__ wl,
                          const float* __restrict__ wr,
                          __bf16* __restrict__ Bpk) {
  int tid = blockIdx.x * blockDim.x + threadIdx.x;
  if (tid >= 8 * 2 * 32) return;
  int lane = tid & 31;
  int tile = tid >> 5;          // tile = kt*8 + nt
  int nt = tile & 7, kt = tile >> 3;
  int n = nt * 16 + (lane & 15);
  __bf16* dstp = Bpk + ((size_t)tile << 9) + lane * 16;
#pragma unroll
  for (int j = 0; j < 16; ++j) {
    int k = kt * 32 + kpos(j, lane);
    float v = (n < 64) ? wl[k * 64 + n] : wr[k * 64 + (n - 64)];
    dstp[j] = (__bf16)v;
  }
}

// Pack W_cat (384 x 512) into B tiles, layout [kt][nt] (Kt=12, Nt=32):
// rows 0-63 Wx[g][0], 64-127 Wx[g][1], 128-255 Wh[g][0], 256-383 Wh[g][1];
// col block g*128..+127 per gate g.
__global__ void pack_wcat(const float* __restrict__ wx,
                          const float* __restrict__ wh,
                          __bf16* __restrict__ Bpk) {
  int tid = blockIdx.x * blockDim.x + threadIdx.x;
  if (tid >= 32 * KT_TOT * 32) return;
  int lane = tid & 31;
  int tile = tid >> 5;          // tile = kt*32 + nt
  int nt = tile & 31, kt = tile >> 5;
  int n = nt * 16 + (lane & 15);
  int g = n >> 7, d = n & 127;
  __bf16* dstp = Bpk + ((size_t)tile << 9) + lane * 16;
#pragma unroll
  for (int j = 0; j < 16; ++j) {
    int k = kt * 32 + kpos(j, lane);
    float v;
    if (k < 64)       v = wx[(((size_t)g * 2 + 0) * 64 + k) * 128 + d];
    else if (k < 128) v = wx[(((size_t)g * 2 + 1) * 64 + (k - 64)) * 128 + d];
    else if (k < 256) v = wh[(((size_t)g * 2 + 0) * 128 + (k - 128)) * 128 + d];
    else              v = wh[(((size_t)g * 2 + 1) * 128 + (k - 256)) * 128 + d];
    dstp[j] = (__bf16)v;
  }
}

__global__ void build_bias(const float* __restrict__ bl, const float* __restrict__ br,
                           const float* __restrict__ bx, const float* __restrict__ bh,
                           const float* __restrict__ bg,
                           float* __restrict__ bgat, float* __restrict__ bgate) {
  int i = blockIdx.x * blockDim.x + threadIdx.x;
  if (i < 128) bgat[i] = (i < 64) ? bl[i] : br[i - 64];
  if (i < 512) {
    int g = i >> 7, d = i & 127;
    bgate[i] = bx[g * 128 + d] + bh[g * 128 + d] + bg[g * 128 + d];
  }
}

__global__ void zero_hc(float* __restrict__ h, float* __restrict__ c) {
  int i = blockIdx.x * blockDim.x + threadIdx.x;
  if (i < N_NODES * H_DIM) { h[i] = 0.0f; c[i] = 0.0f; }
}

__global__ void step_init(float* __restrict__ Tx, float* __restrict__ Th,
                          float* __restrict__ deg, float* __restrict__ sumb,
                          unsigned* __restrict__ maxb) {
  int i = blockIdx.x * blockDim.x + threadIdx.x;
  if (i < N_NODES * F_IN) Tx[i] = 0.0f;
  if (i < N_NODES * H_DIM) Th[i] = 0.0f;
  if (i < N_NODES) { deg[i] = 0.0f; sumb[i] = 0.0f; maxb[i] = 0u; }
}

// ---------- WMMA GEMM --------------------------------------------------------
// C(M x N) = A(bf16,packed) @ B(bf16,packed [kt][nt]) + bias[col].
// Each wave owns one 16-row M-tile and NTW consecutive 16-col N-tiles.
// Per K-step: one A fragment load + NTW contiguous B fragment loads
// (immediate offsets, clause-groupable), then NTW independent WMMAs.
template <int NTW>
__global__ __launch_bounds__(256)
void gemm_bf16_wmma(const __bf16* __restrict__ Apk, int aKtStride, int aKtBase,
                    const __bf16* __restrict__ Bpk, int Kt, int NtTot,
                    const float* __restrict__ bias,
                    float* __restrict__ C, int Mt, int NtGroups, int ldc) {
  int wave = (blockIdx.x * blockDim.x + threadIdx.x) >> 5;
  int lane = threadIdx.x & 31;
  if (wave >= Mt * NtGroups) return;        // wave-uniform: EXEC all-ones below
  int mt = wave / NtGroups;
  int nt0 = (wave % NtGroups) * NTW;
  v8f acc[NTW] = {};
  const __bf16* ap = Apk + (((size_t)(mt * aKtStride + aKtBase)) << 9) + lane * 16;
  const __bf16* bbase = Bpk + ((size_t)nt0 << 9) + lane * 16;
  for (int kt = 0; kt < Kt; ++kt) {
    v16bf a = *(const v16bf*)ap;
    ap += 512;
    const __bf16* bp = bbase + (((size_t)kt * NtTot) << 9);
    v16bf b[NTW];
#pragma unroll
    for (int n = 0; n < NTW; ++n) b[n] = *(const v16bf*)(bp + ((size_t)n << 9));
#pragma unroll
    for (int n = 0; n < NTW; ++n)
      acc[n] = __builtin_amdgcn_wmma_f32_16x16x32_bf16(false, a, false, b[n],
                                                       (short)0, acc[n],
                                                       false, false);
  }
  int row0 = mt * 16 + ((lane >> 4) << 3);  // lanes 16-31 hold M=8..15
  int cl = lane & 15;
#pragma unroll
  for (int n = 0; n < NTW; ++n) {
    int col = (nt0 + n) * 16 + cl;
    float bv = bias[col];
#pragma unroll
    for (int r = 0; r < 8; ++r)
      C[(size_t)(row0 + r) * ldc + col] = acc[n][r] + bv;
  }
}

// ---------- edge kernels (wave32 per edge where rows are gathered) -----------

__global__ void edge_logits(const int* __restrict__ src, const int* __restrict__ dst,
                            const float* __restrict__ XLR, const float* __restrict__ att,
                            float* __restrict__ logit, unsigned* __restrict__ maxb) {
  int e = (blockIdx.x * blockDim.x + threadIdx.x) >> 5;
  int lane = threadIdx.x & 31;
  if (e >= EN_EDGES) return;
  int s, d;
  if (e < E_EDGES) { s = src[e]; d = dst[e]; } else { s = e - E_EDGES; d = s; }
  const float* xl = XLR + (size_t)s * 128;        // lin_l(x)[src]
  const float* xr = XLR + (size_t)d * 128 + 64;   // lin_r(x)[dst]
  float2 a  = *(const float2*)(xl + lane * 2);
  float2 b  = *(const float2*)(xr + lane * 2);
  float2 at = *(const float2*)(att + lane * 2);
  float m0 = a.x + b.x; m0 = (m0 > 0.0f) ? m0 : 0.2f * m0;
  float m1 = a.y + b.y; m1 = (m1 > 0.0f) ? m1 : 0.2f * m1;
  float p = m0 * at.x + m1 * at.y;
#pragma unroll
  for (int off = 16; off; off >>= 1) p += __shfl_xor(p, off, 32);
  if (lane == 0) {
    logit[e] = p;
    atomicMax(maxb + d, ordf(p));
  }
}

__global__ void edge_exp(const int* __restrict__ dst, float* __restrict__ logit,
                         const unsigned* __restrict__ maxb, float* __restrict__ sumb) {
  int e = blockIdx.x * blockDim.x + threadIdx.x;
  if (e >= EN_EDGES) return;
  int d = (e < E_EDGES) ? dst[e] : (e - E_EDGES);
  float ev = __expf(logit[e] - unordf(maxb[d]));
  logit[e] = ev;                 // reuse buffer for exp values
  atomicAdd(sumb + d, ev);
}

__global__ void edge_w(const int* __restrict__ src, const int* __restrict__ dst,
                       const float* __restrict__ ev, const float* __restrict__ sumb,
                       float* __restrict__ deg, float* __restrict__ wbuf) {
  int e = blockIdx.x * blockDim.x + threadIdx.x;
  if (e >= EN_EDGES) return;
  int s, d;
  if (e < E_EDGES) { s = src[e]; d = dst[e]; } else { s = e - E_EDGES; d = s; }
  float alpha = ev[e] / (sumb[d] + 1e-16f);
  float w = (s != d) ? alpha : 0.0f;
  wbuf[e] = w;
  if (w != 0.0f) atomicAdd(deg + s, w);
}

__global__ void node_dis(const float* __restrict__ deg, float* __restrict__ dis) {
  int n = blockIdx.x * blockDim.x + threadIdx.x;
  if (n >= N_NODES) return;
  float dv = deg[n];
  dis[n] = (dv > 0.0f) ? rsqrtf(dv) : 0.0f;
}

// SpMM: Tx[dst] -= nw*xt[src] (64 feat), Th[dst] -= nw*h[src] (128 feat)
__global__ void spmm_edges(const int* __restrict__ src, const int* __restrict__ dst,
                           const float* __restrict__ wbuf, const float* __restrict__ dis,
                           const float* __restrict__ xt, const float* __restrict__ h,
                           float* __restrict__ Tx, float* __restrict__ Th) {
  int e = (blockIdx.x * blockDim.x + threadIdx.x) >> 5;
  int lane = threadIdx.x & 31;
  if (e >= EN_EDGES) return;
  int s, d;
  if (e < E_EDGES) { s = src[e]; d = dst[e]; } else { s = e - E_EDGES; d = s; }
  float w = wbuf[e];
  if (w == 0.0f) return;                     // wave-uniform skip (self loops)
  float coef = -(dis[s] * w * dis[d]);
  float2 xv = *(const float2*)(xt + (size_t)s * F_IN + lane * 2);
  atomicAdd(Tx + (size_t)d * F_IN + lane * 2 + 0, coef * xv.x);
  atomicAdd(Tx + (size_t)d * F_IN + lane * 2 + 1, coef * xv.y);
  float4 hv = *(const float4*)(h + (size_t)s * H_DIM + lane * 4);
  atomicAdd(Th + (size_t)d * H_DIM + lane * 4 + 0, coef * hv.x);
  atomicAdd(Th + (size_t)d * H_DIM + lane * 4 + 1, coef * hv.y);
  atomicAdd(Th + (size_t)d * H_DIM + lane * 4 + 2, coef * hv.z);
  atomicAdd(Th + (size_t)d * H_DIM + lane * 4 + 3, coef * hv.w);
}

// ---------- LSTM pointwise + output ------------------------------------------

__global__ void lstm_update(const float* __restrict__ G, const float* __restrict__ wc,
                            float* __restrict__ h, float* __restrict__ c) {
  int i = blockIdx.x * blockDim.x + threadIdx.x;
  if (i >= N_NODES * H_DIM) return;
  int n = i >> 7, d = i & 127;
  const float* g = G + (size_t)n * 512;
  float cold = c[i];
  float ig = sigmoidf(g[d]        + wc[d]        * cold);
  float fg = sigmoidf(g[128 + d]  + wc[128 + d]  * cold);
  float tg = tanhf(g[256 + d]);
  float cn = fg * cold + ig * tg;
  float og = sigmoidf(g[384 + d]  + wc[256 + d]  * cn);
  c[i] = cn;
  h[i] = og * tanhf(cn);
}

__global__ void final_out(const float* __restrict__ h, const float* __restrict__ lw,
                          const float* __restrict__ lb, float* __restrict__ out) {
  int node = (blockIdx.x * blockDim.x + threadIdx.x) >> 5;
  int lane = threadIdx.x & 31;
  if (node >= N_NODES) return;
  float4 hv = *(const float4*)(h + (size_t)node * H_DIM + lane * 4);
  float4 wv = *(const float4*)(lw + lane * 4);
  float p = hv.x * wv.x + hv.y * wv.y + hv.z * wv.z + hv.w * wv.w;
#pragma unroll
  for (int off = 16; off; off >>= 1) p += __shfl_xor(p, off, 32);
  if (lane == 0) out[node] = p + lb[0];
}

__global__ void copy_hc(const float* __restrict__ h, const float* __restrict__ c,
                        float* __restrict__ out) {
  int i = blockIdx.x * blockDim.x + threadIdx.x;
  if (i >= N_NODES * H_DIM) return;
  out[N_NODES + i] = h[i];
  out[N_NODES + N_NODES * H_DIM + i] = c[i];
}

// ---------- host launcher ----------------------------------------------------

extern "C" void kernel_launch(void* const* d_in, const int* in_sizes, int n_in,
                              void* d_out, int out_size, void* d_ws, size_t ws_size,
                              hipStream_t stream) {
  const float* x   = (const float*)d_in[0];
  const int*   ei  = (const int*)d_in[1];
  // d_in[2] = edge_weight: unused by the reference forward pass
  const float* wl  = (const float*)d_in[3];
  const float* bl  = (const float*)d_in[4];
  const float* wr  = (const float*)d_in[5];
  const float* br  = (const float*)d_in[6];
  const float* att = (const float*)d_in[7];
  const float* wx  = (const float*)d_in[8];
  const float* bx  = (const float*)d_in[9];
  const float* wh  = (const float*)d_in[10];
  const float* bh  = (const float*)d_in[11];
  const float* wc  = (const float*)d_in[12];
  const float* bg  = (const float*)d_in[13];
  const float* lw  = (const float*)d_in[14];
  const float* lb  = (const float*)d_in[15];
  float* out = (float*)d_out;

  char* ws = (char*)d_ws;
  size_t off = 0;
  auto alloc = [&](size_t bytes) -> void* {
    off = (off + 255) & ~(size_t)255;
    void* p = ws + off;
    off += bytes;
    return p;
  };

  float*    h     = (float*)alloc((size_t)N_NODES * H_DIM * 4);
  float*    c     = (float*)alloc((size_t)N_NODES * H_DIM * 4);
  float*    Gates = (float*)alloc((size_t)N_NODES * 512 * 4);
  float*    XLR   = Gates;   // disjoint lifetime within a step: alias saves ws
  float*    Tx    = (float*)alloc((size_t)N_NODES * F_IN * 4);
  float*    Th    = (float*)alloc((size_t)N_NODES * H_DIM * 4);
  float*    logit = (float*)alloc((size_t)EN_EDGES * 4);
  float*    wbuf  = (float*)alloc((size_t)EN_EDGES * 4);
  float*    deg   = (float*)alloc((size_t)N_NODES * 4);
  float*    sumb  = (float*)alloc((size_t)N_NODES * 4);
  float*    dis   = (float*)alloc((size_t)N_NODES * 4);
  unsigned* maxb  = (unsigned*)alloc((size_t)N_NODES * 4);
  __bf16*   Apk   = (__bf16*)alloc((size_t)MT * KT_TOT * 512 * 2);
  __bf16*   Wg    = (__bf16*)alloc((size_t)8 * 2 * 512 * 2);
  __bf16*   Wcat  = (__bf16*)alloc((size_t)32 * KT_TOT * 512 * 2);
  float*    bgat  = (float*)alloc(128 * 4);
  float*    bgate = (float*)alloc(512 * 4);
  (void)ws_size; (void)in_sizes; (void)n_in; (void)out_size;

  const int TPB = 256;
  const int NH  = N_NODES * H_DIM;

  // One-time per launch (deterministic): pack weights, biases, zero state.
  pack_wgat<<<2, TPB, 0, stream>>>(wl, wr, Wg);
  pack_wcat<<<(32 * KT_TOT * 32 + TPB - 1) / TPB, TPB, 0, stream>>>(wx, wh, Wcat);
  build_bias<<<2, TPB, 0, stream>>>(bl, br, bx, bh, bg, bgat, bgate);
  zero_hc<<<(NH + TPB - 1) / TPB, TPB, 0, stream>>>(h, c);

  for (int t = 0; t < T_STEPS; ++t) {
    const float* xt  = x + (size_t)t * N_NODES * F_IN;
    const int*   src = ei + (size_t)t * 2 * E_EDGES;
    const int*   dst = src + E_EDGES;

    // GAT projections via WMMA: XLR = xt @ [Wl|Wr] + [bl|br]
    pack_a_bf16<<<(MT * 2 * 32 + TPB - 1) / TPB, TPB, 0, stream>>>(
        xt, F_IN, Apk, 0, 2, KT_TOT);
    gemm_bf16_wmma<8><<<(MT * 1 + 7) / 8, TPB, 0, stream>>>(
        Apk, KT_TOT, 0, Wg, 2, 8, bgat, XLR, MT, 1, 128);

    // attention softmax over incoming edges, then normalized laplacian weights
    step_init<<<(NH + TPB - 1) / TPB, TPB, 0, stream>>>(Tx, Th, deg, sumb, maxb);
    edge_logits<<<(EN_EDGES + 7) / 8, TPB, 0, stream>>>(src, dst, XLR, att, logit, maxb);
    edge_exp<<<(EN_EDGES + TPB - 1) / TPB, TPB, 0, stream>>>(dst, logit, maxb, sumb);
    edge_w<<<(EN_EDGES + TPB - 1) / TPB, TPB, 0, stream>>>(src, dst, logit, sumb, deg, wbuf);
    node_dis<<<(N_NODES + TPB - 1) / TPB, TPB, 0, stream>>>(deg, dis);

    // Tx = L_hat @ xt, Th = L_hat @ h (shared across all 4 gates)
    spmm_edges<<<(EN_EDGES + 7) / 8, TPB, 0, stream>>>(src, dst, wbuf, dis, xt, h, Tx, Th);

    // Fused gate GEMM: Gates = [xt|Tx|h|Th] @ Wcat + (bx+bh+b_g)
    pack_a_bf16<<<(MT * 2 * 32 + TPB - 1) / TPB, TPB, 0, stream>>>(
        Tx, F_IN, Apk, 2, 2, KT_TOT);
    pack_a_bf16<<<(MT * 4 * 32 + TPB - 1) / TPB, TPB, 0, stream>>>(
        h, H_DIM, Apk, 4, 4, KT_TOT);
    pack_a_bf16<<<(MT * 4 * 32 + TPB - 1) / TPB, TPB, 0, stream>>>(
        Th, H_DIM, Apk, 8, 4, KT_TOT);
    gemm_bf16_wmma<8><<<(MT * 4 + 7) / 8, TPB, 0, stream>>>(
        Apk, KT_TOT, 0, Wcat, KT_TOT, 32, bgate, Gates, MT, 4, 512);

    lstm_update<<<(NH + TPB - 1) / TPB, TPB, 0, stream>>>(Gates, wc, h, c);
  }

  final_out<<<(N_NODES + 7) / 8, TPB, 0, stream>>>(h, lw, lb, out);
  copy_hc<<<(NH + TPB - 1) / TPB, TPB, 0, stream>>>(h, c, out);
}
